// MSARowAttention_80711025426887
// MI455X (gfx1250) — matrix operand order repaired and checked by reading, not compiled, not run on hardware
//
#include <hip/hip_runtime.h>
#include <hip/hip_bf16.h>
#include <math.h>

// ---------------------------------------------------------------------------
// MSA row attention (tied) for MI455X / gfx1250.
// GEMMs on v_wmma_f32_16x16x32_bf16; tile staging via GLOBAL_LOAD_ASYNC_TO_LDS
// (ASYNCcnt-tracked, double buffered) when the toolchain exposes the builtins.
// ---------------------------------------------------------------------------

typedef __attribute__((ext_vector_type(16))) __bf16 v16bf;
typedef __attribute__((ext_vector_type(8)))  __bf16 bf16x8;
typedef __attribute__((ext_vector_type(8)))  float  v8f;
typedef __attribute__((ext_vector_type(4)))  int    v4i;

#define LN_EPS 1e-5f
#define Dd   128
#define Ld   384
#define Cd   256
#define CBd  256
#define Gd   8
#define PPGd 32
#define Mrows (Dd * Ld)          // 49152

#if __has_builtin(__builtin_amdgcn_global_load_async_to_lds_b128) && \
    __has_builtin(__builtin_amdgcn_s_wait_asynccnt)
#define HAVE_ASYNC 1
#else
#define HAVE_ASYNC 0
#if defined(__AMDGCN__)
#warning "gfx1250 async-to-LDS builtins unavailable on device pass; synchronous staging"
#endif
#endif

// ---------------- wave reductions (wave32) ----------------
static __device__ __forceinline__ float wsum(float v) {
#pragma unroll
  for (int m = 16; m >= 1; m >>= 1) v += __shfl_xor(v, m, 32);
  return v;
}
static __device__ __forceinline__ float wmax(float v) {
#pragma unroll
  for (int m = 16; m >= 1; m >>= 1) v = fmaxf(v, __shfl_xor(v, m, 32));
  return v;
}

// ---------------- async 32-byte tile staging ----------------
// INST_OFFSET is added to BOTH the global and LDS addresses (ISA 10.x), so one
// (gsrc,ldst) pair stages 32 contiguous bytes with two b128 asyncs.
static __device__ __forceinline__ void stage32B(const __bf16* gsrc, __bf16* ldst) {
#if HAVE_ASYNC
  __builtin_amdgcn_global_load_async_to_lds_b128(
      (__attribute__((address_space(1))) v4i*)gsrc,
      (__attribute__((address_space(3))) v4i*)ldst, 0, 0);
  __builtin_amdgcn_global_load_async_to_lds_b128(
      (__attribute__((address_space(1))) v4i*)gsrc,
      (__attribute__((address_space(3))) v4i*)ldst, 16, 0);
#else
  *(v16bf*)ldst = *(const v16bf*)gsrc;
#endif
}
static __device__ __forceinline__ void async_fence() {
#if HAVE_ASYNC
  __builtin_amdgcn_s_wait_asynccnt(0);   // s_wait_asynccnt 0
#endif
  __syncthreads();
}

// ---------------- WMMA fragment loads from LDS tiles ----------------
// A fragment (16x32 bf16, MxK): lane 0-15 -> M=lane,   K = {0..7, 16..23}
//                               lane16-31 -> M=lane-16,K = {8..15,24..31}
static __device__ __forceinline__ v16bf load_a_frag(const __bf16* tile, int mt, int lane) {
  const int row = mt + (lane & 15);
  const int kb  = (lane >> 4) << 3;            // 0 or 8
  bf16x8 lo = *(const bf16x8*)(tile + row * 32 + kb);
  bf16x8 hi = *(const bf16x8*)(tile + row * 32 + kb + 16);
  v16bf a;
#pragma unroll
  for (int i = 0; i < 8; ++i) { a[i] = lo[i]; a[i + 8] = hi[i]; }
  return a;
}
// B fragment (32x16 bf16, KxN): lane 0-15 -> N=lane, K={0..7,16..23} ; lane16-31 -> K={8..15,24..31}
// Tile stored as [col][k] (32 k's contiguous per column).
static __device__ __forceinline__ v16bf load_b_frag(const __bf16* tile, int nt, int lane) {
  const int col = nt + (lane & 15);
  const int kb  = (lane >> 4) << 3;            // 0 or 8
  bf16x8 lo = *(const bf16x8*)(tile + col * 32 + kb);
  bf16x8 hi = *(const bf16x8*)(tile + col * 32 + kb + 16);
  v16bf b;
#pragma unroll
  for (int i = 0; i < 8; ++i) { b[i] = lo[i]; b[i + 8] = hi[i]; }
  return b;
}
static __device__ __forceinline__ v8f wmma_bf16(v16bf a, v16bf b, v8f c) {
  return __builtin_amdgcn_wmma_f32_16x16x32_bf16(false, a, false, b, (short)0, c,
                                                 false, false);
}

// ---------------- kernel 1: weights -> bf16 (scales folded) ----------------
__global__ void __launch_bounds__(256)
k_w2bf(const float* __restrict__ Wq, const float* __restrict__ Wk,
       const float* __restrict__ Wv, const float* __restrict__ Wg,
       const float* __restrict__ Wo, __bf16* __restrict__ Wall,
       __bf16* __restrict__ Wobf) {
  const int idx = blockIdx.x * 256 + threadIdx.x;
  if (idx < 1024 * 256) {
    const int j = idx >> 8, k = idx & 255;
    const int which = j >> 8, c = j & 255;
    const float* W;  float s;
    if (which == 0)      { W = Wq; s = 1.0f / 128.0f; }       // q / D
    else if (which == 1) { W = Wk; s = 0.17677669529f; }      // k / sqrt(PPG)
    else if (which == 2) { W = Wv; s = 1.0f; }
    else                 { W = Wg; s = 1.0f; }
    Wall[idx] = (__bf16)(W[c * 256 + k] * s);
  } else {
    const int i2 = idx - 1024 * 256;
    if (i2 < 256 * 256) Wobf[i2] = (__bf16)Wo[i2];
  }
}

// ---------------- kernel 2: pair-bias LN + project to G groups -------------
__global__ void __launch_bounds__(256)
k_pairbias(const float* __restrict__ bias, const float* __restrict__ Wb,
           const float* __restrict__ g_lnb, const float* __restrict__ b_lnb,
           float* __restrict__ pairb) {
  __shared__ float sWb[Gd * CBd];
  for (int i = threadIdx.x; i < Gd * CBd; i += 256) sWb[i] = Wb[i];
  __syncthreads();
  const int lane = threadIdx.x & 31;
  const int row  = blockIdx.x * 8 + (threadIdx.x >> 5);   // 0..L*L-1
  const float* src = bias + (size_t)row * CBd + lane * 8;
  float4 a0 = ((const float4*)src)[0];
  float4 a1 = ((const float4*)src)[1];
  float x[8] = {a0.x, a0.y, a0.z, a0.w, a1.x, a1.y, a1.z, a1.w};
  float s = 0.f, s2 = 0.f;
#pragma unroll
  for (int i = 0; i < 8; ++i) { s += x[i]; s2 += x[i] * x[i]; }
  s = wsum(s); s2 = wsum(s2);
  const float mean = s * (1.0f / CBd);
  const float inv  = rsqrtf(s2 * (1.0f / CBd) - mean * mean + LN_EPS);
  float4 gg0 = ((const float4*)(g_lnb + lane * 8))[0];
  float4 gg1 = ((const float4*)(g_lnb + lane * 8))[1];
  float4 bb0 = ((const float4*)(b_lnb + lane * 8))[0];
  float4 bb1 = ((const float4*)(b_lnb + lane * 8))[1];
  float gv[8] = {gg0.x, gg0.y, gg0.z, gg0.w, gg1.x, gg1.y, gg1.z, gg1.w};
  float bv[8] = {bb0.x, bb0.y, bb0.z, bb0.w, bb1.x, bb1.y, bb1.z, bb1.w};
  float xn[8];
#pragma unroll
  for (int i = 0; i < 8; ++i) xn[i] = (x[i] - mean) * inv * gv[i] + bv[i];
#pragma unroll
  for (int g = 0; g < Gd; ++g) {
    float p = 0.f;
#pragma unroll
    for (int i = 0; i < 8; ++i) p += xn[i] * sWb[g * CBd + lane * 8 + i];
    p = wsum(p);
    if (lane == 0) pairb[(size_t)row * Gd + g] = p;
  }
}

// ---------------- kernel 3: LN(x) -> bf16 ----------------------------------
__global__ void __launch_bounds__(256)
k_lnx(const float* __restrict__ x, const float* __restrict__ g_ln,
      const float* __restrict__ b_ln, __bf16* __restrict__ xnbf) {
  const int lane = threadIdx.x & 31;
  const int row  = blockIdx.x * 8 + (threadIdx.x >> 5);   // 0..Mrows-1
  const float* src = x + (size_t)row * Cd + lane * 8;
  float4 a0 = ((const float4*)src)[0];
  float4 a1 = ((const float4*)src)[1];
  float v[8] = {a0.x, a0.y, a0.z, a0.w, a1.x, a1.y, a1.z, a1.w};
  float s = 0.f, s2 = 0.f;
#pragma unroll
  for (int i = 0; i < 8; ++i) { s += v[i]; s2 += v[i] * v[i]; }
  s = wsum(s); s2 = wsum(s2);
  const float mean = s * (1.0f / Cd);
  const float inv  = rsqrtf(s2 * (1.0f / Cd) - mean * mean + LN_EPS);
  float4 gg0 = ((const float4*)(g_ln + lane * 8))[0];
  float4 gg1 = ((const float4*)(g_ln + lane * 8))[1];
  float4 bb0 = ((const float4*)(b_ln + lane * 8))[0];
  float4 bb1 = ((const float4*)(b_ln + lane * 8))[1];
  float gv[8] = {gg0.x, gg0.y, gg0.z, gg0.w, gg1.x, gg1.y, gg1.z, gg1.w};
  float bv[8] = {bb0.x, bb0.y, bb0.z, bb0.w, bb1.x, bb1.y, bb1.z, bb1.w};
  bf16x8 o;
#pragma unroll
  for (int i = 0; i < 8; ++i) o[i] = (__bf16)((v[i] - mean) * inv * gv[i] + bv[i]);
  *(bf16x8*)(xnbf + (size_t)row * Cd + lane * 8) = o;
}

// ---------------- kernel 4: fused [q|k|v|gate] GEMM -------------------------
// out = xn @ Wall^T ; 128x128 tile/block, 8 waves (4Mx2N), double-buffered LDS.
__global__ void __launch_bounds__(256)
k_qkvg(const __bf16* __restrict__ xnbf, const __bf16* __restrict__ Wall,
       const float* __restrict__ bg, __bf16* __restrict__ qT,
       __bf16* __restrict__ kT, __bf16* __restrict__ vT2,
       __bf16* __restrict__ gateb) {
  __shared__ __align__(32) __bf16 At[2][128 * 32];
  __shared__ __align__(32) __bf16 Bt[2][128 * 32];
  const int tid = threadIdx.x, lane = tid & 31, w = tid >> 5;
  const int wm = w >> 1, wn = w & 1;
  const int m0 = blockIdx.x * 128, n0 = blockIdx.y * 128;
  const v8f vzero = {0.f, 0.f, 0.f, 0.f, 0.f, 0.f, 0.f, 0.f};
  v8f acc[2][4];
#pragma unroll
  for (int i = 0; i < 2; ++i)
#pragma unroll
    for (int j = 0; j < 4; ++j) acc[i][j] = vzero;

  const int r = tid >> 1, half = tid & 1;
  const __bf16* aSrc = xnbf + (size_t)(m0 + r) * Cd + half * 16;
  const __bf16* bSrc = Wall + (size_t)(n0 + r) * Cd + half * 16;
  __bf16* aDst = &At[0][r * 32 + half * 16];
  __bf16* bDst = &Bt[0][r * 32 + half * 16];

  stage32B(aSrc, aDst);
  stage32B(bSrc, bDst);
  async_fence();
  int p = 0;
  for (int k0 = 0; k0 < Cd; k0 += 32) {
    const int nk = k0 + 32;
    if (nk < Cd) {                       // prefetch next tile into other buffer
      stage32B(aSrc + nk, aDst + (p ^ 1) * 4096);
      stage32B(bSrc + nk, bDst + (p ^ 1) * 4096);
    }
    v16bf af[2], bfr[4];
#pragma unroll
    for (int i = 0; i < 2; ++i) af[i] = load_a_frag(At[p], wm * 32 + i * 16, lane);
#pragma unroll
    for (int j = 0; j < 4; ++j) bfr[j] = load_b_frag(Bt[p], wn * 64 + j * 16, lane);
#pragma unroll
    for (int i = 0; i < 2; ++i)
#pragma unroll
      for (int j = 0; j < 4; ++j) acc[i][j] = wmma_bf16(af[i], bfr[j], acc[i][j]);
    async_fence();
    p ^= 1;
  }
  // epilogue: scatter into attention-friendly layouts
#pragma unroll
  for (int i = 0; i < 2; ++i)
#pragma unroll
    for (int j = 0; j < 4; ++j)
#pragma unroll
      for (int e = 0; e < 8; ++e) {
        const int ml  = ((lane >> 4) << 3) + e;
        const int row = m0 + wm * 32 + i * 16 + ml;
        const int col = n0 + wn * 64 + j * 16 + (lane & 15);
        const float a = acc[i][j][e];
        const int d = row / Ld, l = row - d * Ld;
        const int which = col >> 8, c = col & 255;
        const int g = c & 7, pg = c >> 3;
        if (which == 0)
          qT[((size_t)(g * Ld + l)) * 4096 + d * PPGd + pg] = (__bf16)a;
        else if (which == 1)
          kT[((size_t)(g * Ld + l)) * 4096 + d * PPGd + pg] = (__bf16)a;
        else if (which == 2)
          vT2[((size_t)((d * Gd + g) * PPGd + pg)) * Ld + l] = (__bf16)a;
        else {
          const float sg = 1.0f / (1.0f + __expf(-(a + bg[c])));
          gateb[((size_t)((d * Gd + g) * Ld + l)) * PPGd + pg] = (__bf16)sg;
        }
      }
}

// ---------------- kernel 5: tied logits GEMM per group ----------------------
// logits[g][l][m] = sum_{d,pg} q * k + pairb ; K = 4096, double buffered.
__global__ void __launch_bounds__(256)
k_logits(const __bf16* __restrict__ qT, const __bf16* __restrict__ kT,
         const float* __restrict__ pairb, float* __restrict__ logits) {
  __shared__ __align__(32) __bf16 At[2][128 * 32];
  __shared__ __align__(32) __bf16 Bt[2][128 * 32];
  const int tid = threadIdx.x, lane = tid & 31, w = tid >> 5;
  const int wm = w >> 1, wn = w & 1;
  const int g  = blockIdx.y;
  const int l0 = (blockIdx.x / 3) * 128, mm0 = (blockIdx.x % 3) * 128;
  const v8f vzero = {0.f, 0.f, 0.f, 0.f, 0.f, 0.f, 0.f, 0.f};
  v8f acc[2][4];
#pragma unroll
  for (int i = 0; i < 2; ++i)
#pragma unroll
    for (int j = 0; j < 4; ++j) acc[i][j] = vzero;

  const int r = tid >> 1, half = tid & 1;
  const __bf16* aSrc = qT + (size_t)(g * Ld + l0 + r) * 4096 + half * 16;
  const __bf16* bSrc = kT + (size_t)(g * Ld + mm0 + r) * 4096 + half * 16;
  __bf16* aDst = &At[0][r * 32 + half * 16];
  __bf16* bDst = &Bt[0][r * 32 + half * 16];

  stage32B(aSrc, aDst);
  stage32B(bSrc, bDst);
  async_fence();
  int p = 0;
  for (int k0 = 0; k0 < 4096; k0 += 32) {
    const int nk = k0 + 32;
    if (nk < 4096) {
      stage32B(aSrc + nk, aDst + (p ^ 1) * 4096);
      stage32B(bSrc + nk, bDst + (p ^ 1) * 4096);
    }
    v16bf af[2], bfr[4];
#pragma unroll
    for (int i = 0; i < 2; ++i) af[i] = load_a_frag(At[p], wm * 32 + i * 16, lane);
#pragma unroll
    for (int j = 0; j < 4; ++j) bfr[j] = load_b_frag(Bt[p], wn * 64 + j * 16, lane);
#pragma unroll
    for (int i = 0; i < 2; ++i)
#pragma unroll
      for (int j = 0; j < 4; ++j) acc[i][j] = wmma_bf16(af[i], bfr[j], acc[i][j]);
    async_fence();
    p ^= 1;
  }
#pragma unroll
  for (int i = 0; i < 2; ++i)
#pragma unroll
    for (int j = 0; j < 4; ++j)
#pragma unroll
      for (int e = 0; e < 8; ++e) {
        const int ml = ((lane >> 4) << 3) + e;
        const int l  = l0 + wm * 32 + i * 16 + ml;
        const int m  = mm0 + wn * 64 + j * 16 + (lane & 15);
        logits[((size_t)(g * Ld + l)) * Ld + m] =
            acc[i][j][e] + pairb[((size_t)(l * Ld + m)) * Gd + g];
      }
}

// ---------------- kernel 6: softmax over m ---------------------------------
__global__ void __launch_bounds__(256)
k_softmax(const float* __restrict__ logits, __bf16* __restrict__ attnb) {
  const int lane = threadIdx.x & 31;
  const int row  = blockIdx.x * 8 + (threadIdx.x >> 5);   // g*L + l
  const float* src = logits + (size_t)row * Ld;
  float v[12];
#pragma unroll
  for (int i = 0; i < 12; ++i) v[i] = src[lane + i * 32];
  float mx = -3.4e38f;
#pragma unroll
  for (int i = 0; i < 12; ++i) mx = fmaxf(mx, v[i]);
  mx = wmax(mx);
  float s = 0.f;
#pragma unroll
  for (int i = 0; i < 12; ++i) { v[i] = __expf(v[i] - mx); s += v[i]; }
  s = wsum(s);
  const float inv = 1.0f / s;
  __bf16* dst = attnb + (size_t)row * Ld;
#pragma unroll
  for (int i = 0; i < 12; ++i) dst[lane + i * 32] = (__bf16)(v[i] * inv);
}

// ---------------- kernel 7: o = gate * (attn @ v), 4 depth slices / block ---
// block: 128 l-rows x 32 pg-cols x 4 d ; wave = 16 l-rows, 8 WMMAs per k-step.
__global__ void __launch_bounds__(256)
k_av(const __bf16* __restrict__ attnb, const __bf16* __restrict__ vT2,
     const __bf16* __restrict__ gateb, __bf16* __restrict__ ob) {
  __shared__ __align__(32) __bf16 At[2][128 * 32];
  __shared__ __align__(32) __bf16 Bt[2][4][32 * 32];
  const int tid = threadIdx.x, lane = tid & 31, w = tid >> 5;
  const int l0 = blockIdx.x * 128;
  const int d0 = blockIdx.y * 4, g = blockIdx.z;
  const v8f vzero = {0.f, 0.f, 0.f, 0.f, 0.f, 0.f, 0.f, 0.f};
  v8f acc[4][2];
#pragma unroll
  for (int dd = 0; dd < 4; ++dd) { acc[dd][0] = vzero; acc[dd][1] = vzero; }

  const int r = tid >> 1, half = tid & 1;
  const __bf16* aSrc = attnb + (size_t)(g * Ld + l0 + r) * Ld + half * 16;
  const int dd_s = tid >> 6, pg_s = (tid >> 1) & 31, part_s = tid & 1;
  const __bf16* bSrc =
      vT2 + ((size_t)((d0 + dd_s) * Gd + g) * PPGd + pg_s) * Ld + part_s * 16;
  __bf16* aDst = &At[0][r * 32 + half * 16];
  __bf16* bDst = &Bt[0][dd_s][pg_s * 32 + part_s * 16];

  stage32B(aSrc, aDst);
  stage32B(bSrc, bDst);
  async_fence();
  int p = 0;
  for (int k0 = 0; k0 < Ld; k0 += 32) {
    const int nk = k0 + 32;
    if (nk < Ld) {
      stage32B(aSrc + nk, aDst + (p ^ 1) * 4096);
      stage32B(bSrc + nk, bDst + (p ^ 1) * 4096);   // Bt buffer stride = 4*1024
    }
    v16bf af = load_a_frag(At[p], w * 16, lane);
#pragma unroll
    for (int dd = 0; dd < 4; ++dd)
#pragma unroll
      for (int j = 0; j < 2; ++j) {
        v16bf bfr = load_b_frag(Bt[p][dd], j * 16, lane);
        acc[dd][j] = wmma_bf16(af, bfr, acc[dd][j]);
      }
    async_fence();
    p ^= 1;
  }
#pragma unroll
  for (int dd = 0; dd < 4; ++dd)
#pragma unroll
    for (int j = 0; j < 2; ++j)
#pragma unroll
      for (int e = 0; e < 8; ++e) {
        const int ml = ((lane >> 4) << 3) + e;
        const int l  = l0 + w * 16 + ml;
        const int pg = j * 16 + (lane & 15);
        const int d  = d0 + dd;
        const float gv =
            (float)gateb[((size_t)((d * Gd + g) * Ld + l)) * PPGd + pg];
        ob[((size_t)(d * Ld + l)) * Cd + pg * Gd + g] = (__bf16)(acc[dd][j][e] * gv);
      }
}

// ---------------- kernel 8: output projection + bias + residual -------------
__global__ void __launch_bounds__(256)
k_oproj(const __bf16* __restrict__ ob, const __bf16* __restrict__ Wobf,
        const float* __restrict__ bo, const float* __restrict__ x,
        float* __restrict__ out) {
  __shared__ __align__(32) __bf16 At[2][128 * 32];
  __shared__ __align__(32) __bf16 Bt[2][128 * 32];
  const int tid = threadIdx.x, lane = tid & 31, w = tid >> 5;
  const int wm = w >> 1, wn = w & 1;
  const int m0 = blockIdx.x * 128, n0 = blockIdx.y * 128;
  const v8f vzero = {0.f, 0.f, 0.f, 0.f, 0.f, 0.f, 0.f, 0.f};
  v8f acc[2][4];
#pragma unroll
  for (int i = 0; i < 2; ++i)
#pragma unroll
    for (int j = 0; j < 4; ++j) acc[i][j] = vzero;

  const int r = tid >> 1, half = tid & 1;
  const __bf16* aSrc = ob + (size_t)(m0 + r) * Cd + half * 16;
  const __bf16* bSrc = Wobf + (size_t)(n0 + r) * Cd + half * 16;
  __bf16* aDst = &At[0][r * 32 + half * 16];
  __bf16* bDst = &Bt[0][r * 32 + half * 16];

  stage32B(aSrc, aDst);
  stage32B(bSrc, bDst);
  async_fence();
  int p = 0;
  for (int k0 = 0; k0 < Cd; k0 += 32) {
    const int nk = k0 + 32;
    if (nk < Cd) {
      stage32B(aSrc + nk, aDst + (p ^ 1) * 4096);
      stage32B(bSrc + nk, bDst + (p ^ 1) * 4096);
    }
    v16bf af[2], bfr[4];
#pragma unroll
    for (int i = 0; i < 2; ++i) af[i] = load_a_frag(At[p], wm * 32 + i * 16, lane);
#pragma unroll
    for (int j = 0; j < 4; ++j) bfr[j] = load_b_frag(Bt[p], wn * 64 + j * 16, lane);
#pragma unroll
    for (int i = 0; i < 2; ++i)
#pragma unroll
      for (int j = 0; j < 4; ++j) acc[i][j] = wmma_bf16(af[i], bfr[j], acc[i][j]);
    async_fence();
    p ^= 1;
  }
#pragma unroll
  for (int i = 0; i < 2; ++i)
#pragma unroll
    for (int j = 0; j < 4; ++j)
#pragma unroll
      for (int e = 0; e < 8; ++e) {
        const int ml  = ((lane >> 4) << 3) + e;
        const int row = m0 + wm * 32 + i * 16 + ml;
        const int col = n0 + wn * 64 + j * 16 + (lane & 15);
        out[(size_t)row * Cd + col] =
            acc[i][j][e] + bo[col] + x[(size_t)row * Cd + col];
      }
}

// ---------------------------------------------------------------------------
extern "C" void kernel_launch(void* const* d_in, const int* in_sizes, int n_in,
                              void* d_out, int out_size, void* d_ws, size_t ws_size,
                              hipStream_t stream) {
  const float* x     = (const float*)d_in[0];
  const float* bias  = (const float*)d_in[1];
  const float* Wq    = (const float*)d_in[2];
  const float* Wk    = (const float*)d_in[3];
  const float* Wv    = (const float*)d_in[4];
  const float* Wb    = (const float*)d_in[5];
  const float* Wo    = (const float*)d_in[6];
  const float* bo    = (const float*)d_in[7];
  const float* Wg    = (const float*)d_in[8];
  const float* bg    = (const float*)d_in[9];
  const float* g_ln  = (const float*)d_in[10];
  const float* b_ln  = (const float*)d_in[11];
  const float* g_lnb = (const float*)d_in[12];
  const float* b_lnb = (const float*)d_in[13];
  float* out = (float*)d_out;

  char* ws = (char*)d_ws;
  size_t off = 0;
  auto salloc = [&](size_t bytes) -> char* {
    char* p = ws + off;
    off = (off + bytes + 255) & ~(size_t)255;
    return p;
  };
  __bf16* Wall  = (__bf16*)salloc((size_t)1024 * 256 * 2);
  __bf16* Wobf  = (__bf16*)salloc((size_t)256 * 256 * 2);
  __bf16* xnbf  = (__bf16*)salloc((size_t)Mrows * Cd * 2);       // reused as `ob`
  __bf16* qT    = (__bf16*)salloc((size_t)Gd * Ld * 4096 * 2);   // reused as attn
  __bf16* kT    = (__bf16*)salloc((size_t)Gd * Ld * 4096 * 2);
  __bf16* vT2   = (__bf16*)salloc((size_t)Dd * Gd * PPGd * Ld * 2);
  __bf16* gateb = (__bf16*)salloc((size_t)Dd * Gd * Ld * PPGd * 2);
  float*  pairb = (float*)salloc((size_t)Ld * Ld * Gd * 4);
  float*  logit = (float*)salloc((size_t)Gd * Ld * Ld * 4);
  __bf16* attnb = qT;      // qT dead after k_logits
  __bf16* ob    = xnbf;    // xnbf dead after k_qkvg

  k_w2bf<<<dim3(1280), dim3(256), 0, stream>>>(Wq, Wk, Wv, Wg, Wo, Wall, Wobf);
  k_pairbias<<<dim3((Ld * Ld) / 8), dim3(256), 0, stream>>>(bias, Wb, g_lnb, b_lnb, pairb);
  k_lnx<<<dim3(Mrows / 8), dim3(256), 0, stream>>>(x, g_ln, b_ln, xnbf);
  k_qkvg<<<dim3(Mrows / 128, 1024 / 128), dim3(256), 0, stream>>>(xnbf, Wall, bg, qT,
                                                                  kT, vT2, gateb);
  k_logits<<<dim3(9, Gd), dim3(256), 0, stream>>>(qT, kT, pairb, logit);
  k_softmax<<<dim3((Gd * Ld) / 8), dim3(256), 0, stream>>>(logit, attnb);
  k_av<<<dim3(Ld / 128, Dd / 4, Gd), dim3(256), 0, stream>>>(attnb, vT2, gateb, ob);
  k_oproj<<<dim3(Mrows / 128, Cd / 128), dim3(256), 0, stream>>>(ob, Wobf, bo, x, out);
}
